// MHA_pro_29609504539190
// MI455X (gfx1250) — matrix operand order repaired and checked by reading, not compiled, not run on hardware
//
#include <hip/hip_runtime.h>
#include <hip/hip_bf16.h>
#include <math.h>

// ---------------------------------------------------------------------------
// Problem constants (match reference)
// ---------------------------------------------------------------------------
#define Bc   256
#define Tc   200
#define Cc   512
#define Hc   8
#define HEADc 64
#define ROTc 32
#define TPc  224          // T padded to multiple of 32 for WMMA K-steps
#define MTc  (Bc * Tc)    // 51200 rows

typedef __bf16 bf16_t;
typedef __attribute__((ext_vector_type(16))) __bf16 bf16x16;
typedef __attribute__((ext_vector_type(8)))  float  floatx8;

// Load a 16-element bf16 WMMA operand fragment from two contiguous 8-element
// runs (p0 = first run, p1 = second run). Used for both A and B operands.
__device__ __forceinline__ bf16x16 load_frag16(const bf16_t* p0, const bf16_t* p1) {
  union U { uint4 v; bf16_t h[8]; } lo, hi;
  lo.v = *reinterpret_cast<const uint4*>(p0);
  hi.v = *reinterpret_cast<const uint4*>(p1);
  bf16x16 r;
#pragma unroll
  for (int i = 0; i < 8; ++i) { r[i] = lo.h[i]; r[8 + i] = hi.h[i]; }
  return r;
}

// ---------------------------------------------------------------------------
// 1. fp32 -> bf16 cast (for weight matrices)
// ---------------------------------------------------------------------------
__global__ void cast_f32_bf16(const float* __restrict__ src,
                              bf16_t* __restrict__ dst, int n) {
  for (int i = blockIdx.x * blockDim.x + threadIdx.x; i < n;
       i += gridDim.x * blockDim.x)
    dst[i] = (bf16_t)src[i];
}

// ---------------------------------------------------------------------------
// 2. x = concat(shift(item_emb[log_seqs][:, :, :256]), item_emb[log_seqs][:, :, 256:])
// ---------------------------------------------------------------------------
__global__ __launch_bounds__(256) void gather_shift_bf16(
    const int* __restrict__ log_seqs, const float* __restrict__ item_emb,
    bf16_t* __restrict__ xout) {
  const int bt = blockIdx.x;              // 0..B*T-1
  const int b = bt / Tc, t = bt % Tc;
  for (int c = threadIdx.x; c < Cc; c += blockDim.x) {
    float v;
    if (c < Cc / 2) {
      v = (t == 0) ? 0.f
                   : item_emb[(size_t)log_seqs[b * Tc + t - 1] * Cc + c];
    } else {
      v = item_emb[(size_t)log_seqs[b * Tc + t] * Cc + c];
    }
    xout[(size_t)bt * Cc + c] = (bf16_t)v;
  }
}

// ---------------------------------------------------------------------------
// 3. GEMM: C[M,N] = A[M,K] @ W[N,K]^T + bias, bf16 in, fp32 out.
//    Block = 256 threads = 8 waves. The block's 64-column weight panel
//    (64 x K bf16 = 64 KB) is staged once in LDS (dynamic) and all B
//    fragments come from ds_load_b128. Each wave computes a 32x64 tile
//    (2 A fragments, 8 accumulators -> 8 WMMA per k-step, 2x B reuse).
// ---------------------------------------------------------------------------
__global__ __launch_bounds__(256) void gemm_bf16_bias(
    const bf16_t* __restrict__ A, const bf16_t* __restrict__ W,
    const float* __restrict__ bias, float* __restrict__ Cout,
    int M, int N, int K) {
  extern __shared__ char smem[];
  bf16_t* bpanel = (bf16_t*)smem;  // [64][K]

  const int lane = threadIdx.x & 31;
  const int wave = threadIdx.x >> 5;
  const int row0 = (blockIdx.x * 8 + wave) * 32;
  const int n0 = blockIdx.y * 64;
  const int am = lane & 15;       // A row within tile / B column within tile
  const int ah = lane >> 4;       // lane half selects K sub-ranges

  // ---- stage weight panel W[n0 .. n0+63][0..K) into LDS (contiguous) ----
  {
    const uint4* src = (const uint4*)(W + (size_t)n0 * K);
    uint4* dst = (uint4*)bpanel;
    const int nvec = 64 * K / 8;  // 8 bf16 per uint4
    for (int i = threadIdx.x; i < nvec; i += 256) dst[i] = src[i];
  }
  __syncthreads();
  if (row0 >= M) return;

  const bf16_t* arow0 = A + (size_t)(row0 + am) * K;
  const bf16_t* arow1 = A + (size_t)(row0 + 16 + am) * K;

  floatx8 acc[2][4] = {};
  for (int k0 = 0; k0 < K; k0 += 32) {
    const bf16_t* pa0 = arow0 + k0 + (ah << 3);
    const bf16_t* pa1 = arow1 + k0 + (ah << 3);
    bf16x16 af0 = load_frag16(pa0, pa0 + 16);
    bf16x16 af1 = load_frag16(pa1, pa1 + 16);
    if (k0 + 32 < K) {
      __builtin_prefetch(arow0 + k0 + 32, 0, 0);  // global_prefetch_b8
      __builtin_prefetch(arow1 + k0 + 32, 0, 0);
    }
#pragma unroll
    for (int tn = 0; tn < 4; ++tn) {
      const bf16_t* pb = bpanel + (size_t)(tn * 16 + am) * K + k0 + (ah << 3);
      bf16x16 bfrag = load_frag16(pb, pb + 16);  // ds_load_b128 x2
      acc[0][tn] = __builtin_amdgcn_wmma_f32_16x16x32_bf16(
          false, af0, false, bfrag, (short)0, acc[0][tn], false, false);
      acc[1][tn] = __builtin_amdgcn_wmma_f32_16x16x32_bf16(
          false, af1, false, bfrag, (short)0, acc[1][tn], false, false);
    }
  }
  const int rb = ah * 8;
#pragma unroll
  for (int rt = 0; rt < 2; ++rt) {
#pragma unroll
    for (int tn = 0; tn < 4; ++tn) {
      const int n = n0 + tn * 16 + am;
      const float bv = bias ? bias[n] : 0.f;
#pragma unroll
      for (int r = 0; r < 8; ++r)
        Cout[(size_t)(row0 + rt * 16 + rb + r) * N + n] = acc[rt][tn][r] + bv;
    }
  }
}

// ---------------------------------------------------------------------------
// 4. RoPE (+ fold 1/sqrt(HEAD) into q), repack to padded bf16 layouts:
//    qbf,kbf: (B,H,TP,64)  row-major (zero-padded rows t>=T)
//    vT:      (B,H,64,TP)  transposed so PV B-operand is contiguous in K(=s)
// ---------------------------------------------------------------------------
__global__ __launch_bounds__(64) void rope_pack(
    const float* __restrict__ qf, const float* __restrict__ kf,
    const float* __restrict__ vf, bf16_t* __restrict__ qbf,
    bf16_t* __restrict__ kbf, bf16_t* __restrict__ vT) {
  const int t = blockIdx.x;     // 0..TP-1
  const int h = blockIdx.y;     // 0..H-1
  const int b = blockIdx.z;     // 0..B-1
  const int d = threadIdx.x;    // 0..63
  const size_t qk_idx = (((size_t)(b * Hc + h)) * TPc + t) * HEADc + d;
  const size_t vt_idx = (((size_t)(b * Hc + h)) * HEADc + d) * TPc + t;
  if (t >= Tc) {
    qbf[qk_idx] = (bf16_t)0.f;
    kbf[qk_idx] = (bf16_t)0.f;
    vT[vt_idx] = (bf16_t)0.f;
    return;
  }
  const size_t src = ((size_t)(b * Tc + t)) * Cc + h * HEADc + d;
  float q = qf[src], k = kf[src], v = vf[src];
  if (d < ROTc) {
    const int i = d & 15;  // frequency index (emb = concat(freqs,freqs))
    const float invf = powf(10000.f, -((float)(2 * i)) / (float)ROTc);
    const float ang = (float)t * invf;
    const float cs = cosf(ang), sn = sinf(ang);
    if (d < 16) {  // rotate_half: out = x*cos - x2*sin
      q = q * cs - qf[src + 16] * sn;
      k = k * cs - kf[src + 16] * sn;
    } else {       // out = x*cos + x1*sin
      q = q * cs + qf[src - 16] * sn;
      k = k * cs + kf[src - 16] * sn;
    }
  }
  q *= 0.125f;  // 1/sqrt(HEAD)
  qbf[qk_idx] = (bf16_t)q;
  kbf[qk_idx] = (bf16_t)k;
  vT[vt_idx] = (bf16_t)v;
}

// ---------------------------------------------------------------------------
// 5. Fused attention. One block per (batch, 16-query tile); 8 waves = 8 heads.
//    Dynamic LDS: sc[8][16][TP] f32 raw/softmaxed scores,
//                 pm[8][16][TP] bf16 head-mixed probabilities,
//                 wmix[64] f32.
// ---------------------------------------------------------------------------
#define SC(hh, m, s) sc[(((hh)*16 + (m)) * TPc) + (s)]
#define PM(hh, m, s) pm[(((hh)*16 + (m)) * TPc) + (s)]

__global__ __launch_bounds__(256) void attn_fused(
    const bf16_t* __restrict__ qbf, const bf16_t* __restrict__ kbf,
    const bf16_t* __restrict__ vT, const float* __restrict__ time_w,
    const float* __restrict__ time_alpha, const float* __restrict__ time_beta,
    const float* __restrict__ Wmix, bf16_t* __restrict__ Obf) {
  extern __shared__ char smem[];
  float* sc = (float*)smem;                                    // 114688 B
  bf16_t* pm = (bf16_t*)(smem + Hc * 16 * TPc * 4);            // 57344 B
  float* wmix_s = (float*)(smem + Hc * 16 * TPc * 4 + Hc * 16 * TPc * 2);

  const int b = blockIdx.y;
  const int t0 = blockIdx.x * 16;
  const int wave = threadIdx.x >> 5;  // head
  const int lane = threadIdx.x & 31;
  const int am = lane & 15;
  const int ah = lane >> 4;
  const int h = wave;

  if (threadIdx.x < Hc * Hc) wmix_s[threadIdx.x] = Wmix[threadIdx.x];

  // ---- S = q @ k^T (per head) ----
  const bf16_t* qb = qbf + (((size_t)(b * Hc + h)) * TPc + t0) * HEADc;
  const bf16_t* kb = kbf + ((size_t)(b * Hc + h)) * TPc * HEADc;
  const bf16_t* pq0 = qb + (size_t)am * HEADc + (ah << 3);        // d-half 0..31
  const bf16_t* pq1 = pq0 + 32;                                   // d-half 32..63
  bf16x16 a0 = load_frag16(pq0, pq0 + 16);
  bf16x16 a1 = load_frag16(pq1, pq1 + 16);
  for (int s0 = 0; s0 < TPc; s0 += 16) {
    const bf16_t* pk = kb + (size_t)(s0 + am) * HEADc + (ah << 3);
    bf16x16 b0 = load_frag16(pk, pk + 16);           // d = 0..31
    bf16x16 b1 = load_frag16(pk + 32, pk + 48);      // d = 32..63
    floatx8 accs = {};
    accs = __builtin_amdgcn_wmma_f32_16x16x32_bf16(false, a0, false, b0,
                                                   (short)0, accs, false, false);
    accs = __builtin_amdgcn_wmma_f32_16x16x32_bf16(false, a1, false, b1,
                                                   (short)0, accs, false, false);
    const int rb = ah * 8;
#pragma unroll
    for (int r = 0; r < 8; ++r) SC(h, rb + r, s0 + am) = accs[r];
  }
  __syncthreads();

  // ---- causal softmax + relative-time weight (one thread per row) ----
  if (threadIdx.x < 128) {
    const int hh = threadIdx.x >> 4;
    const int m = threadIdx.x & 15;
    const int t = t0 + m;
    if (t >= Tc) {
      for (int s = 0; s < TPc; ++s) SC(hh, m, s) = 0.f;
    } else {
      float mx = -3.4e38f;
      for (int s = 0; s <= t; ++s) mx = fmaxf(mx, SC(hh, m, s));
      float sum = 0.f;
      for (int s = 0; s <= t; ++s) {
        const float e = __expf(SC(hh, m, s) - mx);
        SC(hh, m, s) = e;
        sum += e;
      }
      const float inv = 1.f / sum;
      const float beta = time_beta[hh * Tc + t];
      for (int s = 0; s < TPc; ++s) {
        if (s <= t) {
          const float w = time_w[hh * Tc + (199 - (t - s))] *
                          time_alpha[hh * Tc + s] * beta;
          SC(hh, m, s) *= inv * w;
        } else {
          SC(hh, m, s) = 0.f;
        }
      }
    }
  }
  __syncthreads();

  // ---- 8x8 head mix: pm[h] = sum_g Wmix[h][g] * sc[g]  (bf16 for WMMA A) ----
  for (int idx = threadIdx.x; idx < 16 * TPc; idx += 256) {
    const int m = idx / TPc, s = idx % TPc;
    float p[Hc];
#pragma unroll
    for (int g = 0; g < Hc; ++g) p[g] = SC(g, m, s);
#pragma unroll
    for (int hh = 0; hh < Hc; ++hh) {
      float a = 0.f;
#pragma unroll
      for (int g = 0; g < Hc; ++g) a += wmix_s[hh * Hc + g] * p[g];
      PM(hh, m, s) = (bf16_t)a;
    }
  }
  __syncthreads();

  // ---- O = pm @ v (per head); A fragments come from LDS ----
  const bf16_t* vb = vT + ((size_t)(b * Hc + h)) * HEADc * TPc;
  floatx8 acco[4] = {};
  for (int s0 = 0; s0 < TPc; s0 += 32) {
    const bf16_t* pp = &PM(h, am, s0 + (ah << 3));
    bf16x16 ap = load_frag16(pp, pp + 16);  // ds_load path
#pragma unroll
    for (int tn = 0; tn < 4; ++tn) {
      const bf16_t* pv = vb + (size_t)(tn * 16 + am) * TPc + s0 + (ah << 3);
      bf16x16 bv = load_frag16(pv, pv + 16);
      acco[tn] = __builtin_amdgcn_wmma_f32_16x16x32_bf16(
          false, ap, false, bv, (short)0, acco[tn], false, false);
    }
  }
  const int rb = ah * 8;
#pragma unroll
  for (int tn = 0; tn < 4; ++tn) {
#pragma unroll
    for (int r = 0; r < 8; ++r) {
      const int t = t0 + rb + r;
      if (t < Tc)
        Obf[((size_t)(b * Tc + t)) * Cc + h * HEADc + tn * 16 + am] =
            (bf16_t)acco[tn][r];
    }
  }
}

// ---------------------------------------------------------------------------
// 7. Epilogue: lf *= gamma[t]; pos_logits = lf * pos_emb (B,T,C);
//    neg_logits = sum_c lf * neg_emb (B,T).
// ---------------------------------------------------------------------------
__global__ __launch_bounds__(256) void posneg_kernel(
    const float* __restrict__ lf, const float* __restrict__ gamma,
    const float* __restrict__ item_emb, const int* __restrict__ pos_seqs,
    const int* __restrict__ neg_seqs, float* __restrict__ pos_out,
    float* __restrict__ neg_out) {
  const int bt = blockIdx.x;
  const int t = bt % Tc;
  const float g = gamma[t];
  const float* pe = item_emb + (size_t)pos_seqs[bt] * Cc;
  const float* ne = item_emb + (size_t)neg_seqs[bt] * Cc;
  const float* l = lf + (size_t)bt * Cc;
  __shared__ float red[256];
  float partial = 0.f;
  for (int c = threadIdx.x; c < Cc; c += 256) {
    const float v = l[c] * g;
    pos_out[(size_t)bt * Cc + c] = v * pe[c];
    partial += v * ne[c];
  }
  red[threadIdx.x] = partial;
  __syncthreads();
  for (int o = 128; o > 0; o >>= 1) {
    if (threadIdx.x < o) red[threadIdx.x] += red[threadIdx.x + o];
    __syncthreads();
  }
  if (threadIdx.x == 0) neg_out[bt] = red[0];
}

// ---------------------------------------------------------------------------
// Host-side orchestration
// ---------------------------------------------------------------------------
extern "C" void kernel_launch(void* const* d_in, const int* in_sizes, int n_in,
                              void* d_out, int out_size, void* d_ws,
                              size_t ws_size, hipStream_t stream) {
  (void)in_sizes; (void)n_in; (void)out_size; (void)ws_size;
  // inputs (setup_inputs order)
  const int*   log_seqs  = (const int*)d_in[1];
  const int*   pos_seqs  = (const int*)d_in[2];
  const int*   neg_seqs  = (const int*)d_in[3];
  const float* item_emb  = (const float*)d_in[4];
  const float* time_w    = (const float*)d_in[5];
  const float* time_alpha= (const float*)d_in[6];
  const float* time_beta = (const float*)d_in[7];
  const float* time_gamma= (const float*)d_in[8];
  const float* Wq = (const float*)d_in[9];
  const float* bq = (const float*)d_in[10];
  const float* Wk = (const float*)d_in[11];
  const float* bk = (const float*)d_in[12];
  const float* Wv = (const float*)d_in[13];
  const float* bv = (const float*)d_in[14];
  const float* Wmix = (const float*)d_in[15];
  const float* Wo = (const float*)d_in[16];
  const float* bo = (const float*)d_in[17];

  float* pos_out = (float*)d_out;                      // B*T*C
  float* neg_out = (float*)d_out + (size_t)MTc * Cc;   // B*T

  // workspace carve
  size_t off = 0;
  auto carve = [&](size_t bytes) -> void* {
    void* p = (char*)d_ws + off;
    off += (bytes + 255) & ~(size_t)255;
    return p;
  };
  bf16_t* x_bf  = (bf16_t*)carve((size_t)MTc * Cc * 2);
  bf16_t* wq_bf = (bf16_t*)carve((size_t)Cc * Cc * 2);
  bf16_t* wk_bf = (bf16_t*)carve((size_t)Cc * Cc * 2);
  bf16_t* wv_bf = (bf16_t*)carve((size_t)Cc * Cc * 2);
  bf16_t* wo_bf = (bf16_t*)carve((size_t)Cc * Cc * 2);
  float*  qf    = (float*)carve((size_t)MTc * Cc * 4);
  float*  kf    = (float*)carve((size_t)MTc * Cc * 4);
  float*  vf    = (float*)carve((size_t)MTc * Cc * 4);
  bf16_t* qbf   = (bf16_t*)carve((size_t)Bc * Hc * TPc * HEADc * 2);
  bf16_t* kbf   = (bf16_t*)carve((size_t)Bc * Hc * TPc * HEADc * 2);
  bf16_t* vT    = (bf16_t*)carve((size_t)Bc * Hc * HEADc * TPc * 2);
  bf16_t* Obf   = (bf16_t*)carve((size_t)MTc * Cc * 2);
  float*  lf    = (float*)carve((size_t)MTc * Cc * 4);

  // 1. weights -> bf16
  const int wn = Cc * Cc;
  cast_f32_bf16<<<(wn + 255) / 256, 256, 0, stream>>>(Wq, wq_bf, wn);
  cast_f32_bf16<<<(wn + 255) / 256, 256, 0, stream>>>(Wk, wk_bf, wn);
  cast_f32_bf16<<<(wn + 255) / 256, 256, 0, stream>>>(Wv, wv_bf, wn);
  cast_f32_bf16<<<(wn + 255) / 256, 256, 0, stream>>>(Wo, wo_bf, wn);

  // 2. gather + shift + cast
  gather_shift_bf16<<<MTc, 256, 0, stream>>>(log_seqs, item_emb, x_bf);

  // 3. QKV projections (WMMA bf16; 64KB dynamic LDS weight panel per block)
  dim3 ggrid(MTc / 256, Cc / 64);
  const size_t gsmem = (size_t)64 * Cc * 2;  // 65536 B
  gemm_bf16_bias<<<ggrid, 256, gsmem, stream>>>(x_bf, wq_bf, bq, qf, MTc, Cc, Cc);
  gemm_bf16_bias<<<ggrid, 256, gsmem, stream>>>(x_bf, wk_bf, bk, kf, MTc, Cc, Cc);
  gemm_bf16_bias<<<ggrid, 256, gsmem, stream>>>(x_bf, wv_bf, bv, vf, MTc, Cc, Cc);

  // 4. RoPE + repack
  rope_pack<<<dim3(TPc, Hc, Bc), 64, 0, stream>>>(qf, kf, vf, qbf, kbf, vT);

  // 5. fused attention (dynamic LDS: scores f32 + mixed bf16 + Wmix)
  const size_t smem_bytes =
      (size_t)Hc * 16 * TPc * 4 + (size_t)Hc * 16 * TPc * 2 + 256;
  attn_fused<<<dim3((Tc + 15) / 16, Bc), 256, smem_bytes, stream>>>(
      qbf, kbf, vT, time_w, time_alpha, time_beta, Wmix, Obf);

  // 6. output projection
  gemm_bf16_bias<<<ggrid, 256, gsmem, stream>>>(Obf, wo_bf, bo, lf, MTc, Cc, Cc);

  // 7. gamma + pos/neg logits
  posneg_kernel<<<MTc, 256, 0, stream>>>(lf, time_gamma, item_emb, pos_seqs,
                                         neg_seqs, pos_out, neg_out);
}
